// Learned_HighDimBPDecoder_38336878084479
// MI455X (gfx1250) — matrix operand order repaired and checked by reading, not compiled, not run on hardware
//
#include <hip/hip_runtime.h>
#include <hip/hip_bf16.h>
#include <math.h>

// ---------------------------------------------------------------------------
// Learned high-dim BP decoder, CDNA5 (gfx1250), wave32 + WMMA f16->f32,
// async global->LDS staging (ASYNCcnt) with pre-transposed f16 weights.
// M=1024 checks, N=2048 vars, DV=3, DC=6, D=8, B=128, T=5.
// ---------------------------------------------------------------------------

typedef __attribute__((ext_vector_type(16))) _Float16 v16h;
typedef __attribute__((ext_vector_type(8)))  float    v8f;
typedef int v4i __attribute__((vector_size(16)));

union F16x16 { v16h v; unsigned int u[8]; };

#if __has_builtin(__builtin_amdgcn_global_load_async_to_lds_b128)
#define HAVE_ASYNC_LDS 1
#else
#define HAVE_ASYNC_LDS 0
#endif

// 16-byte global -> LDS copy (async path uses ASYNCcnt; fallback via VGPRs).
__device__ __forceinline__ void g2l_b128(const _Float16* g, _Float16* l) {
#if HAVE_ASYNC_LDS
  __builtin_amdgcn_global_load_async_to_lds_b128(
      (__attribute__((address_space(1))) v4i*)(v4i*)g,
      (__attribute__((address_space(3))) v4i*)(v4i*)l, 0, 0);
#else
  *(int4*)l = *(const int4*)g;
#endif
}

__device__ __forceinline__ void wait_async_copies() {
#if HAVE_ASYNC_LDS
#if __has_builtin(__builtin_amdgcn_s_wait_asynccnt)
  __builtin_amdgcn_s_wait_asynccnt(0);
#else
  asm volatile("s_wait_asynccnt 0" ::: "memory");
#endif
#endif
}

// contiguous cooperative copy, 8 f16 (16B) granules; elems % 8 == 0
__device__ __forceinline__ void copy_block(const _Float16* __restrict__ g,
                                           _Float16* __restrict__ l,
                                           int elems, int tid, int nthr) {
  for (int i = tid * 8; i < elems; i += nthr * 8) g2l_b128(g + i, l + i);
}

// A-fragment (16x32 f16, M rows x K): lane L<16 holds row L, K chunks
// {16*half+2v} for v<8 (contiguous, matches load_bt_frag).
__device__ __forceinline__ v16h load_a_frag(const _Float16* base, int stride,
                                            int k0, int lane) {
  const int half = lane >> 4, r = lane & 15;
  const _Float16* row = base + r * stride + k0;
  F16x16 f;
#pragma unroll
  for (int v = 0; v < 8; ++v) {
    const int k = 16 * half + 2 * v;
    f.u[v] = *(const unsigned int*)(row + k);       // 2 packed f16, k even
  }
  return f.v;
}

// B-fragment (32x16 f16, K x N) from a TRANSPOSED LDS image W^T[n][k]
// (stride = padded K). Lanes 0-15 -> K 0-15, lanes 16-31 -> K 16-31.
__device__ __forceinline__ v16h load_bt_frag(const _Float16* baseT, int stride,
                                             int k0, int lane) {
  const int half = lane >> 4, c = lane & 15;
  const _Float16* row = baseT + c * stride + k0 + 16 * half;
  F16x16 f;
#pragma unroll
  for (int v = 0; v < 8; ++v) f.u[v] = *(const unsigned int*)(row + 2 * v);
  return f.v;
}

__device__ __forceinline__ v8f wmma_f16(v16h a, v16h b, v8f c) {
  return __builtin_amdgcn_wmma_f32_16x16x32_f16(false, a, false, b,
                                                (short)0, c, false, false);
}

// Branch-free gelu: native TANH transcendental on CDNA5 if available.
__device__ __forceinline__ float fast_tanh(float x) {
#if __has_builtin(__builtin_amdgcn_tanhf)
  return __builtin_amdgcn_tanhf(x);
#else
  const float e = __expf(2.0f * x);
  return 1.0f - 2.0f / (e + 1.0f);
#endif
}

__device__ __forceinline__ float gelu_fast(float x) {
  const float u = 0.7978845608028654f * (x + 0.044715f * x * x * x);
  return 0.5f * x * (1.0f + fast_tanh(u));
}

// ---------------------------------------------------------------------------
// Weight prep: fp32 -> f16, transposed + padded so per-check/var LDS images
// are contiguous global ranges (enables b128 async staging).
// ---------------------------------------------------------------------------
__global__ void prep_w1c(const float* __restrict__ s, _Float16* __restrict__ d) {
  // d[m][h=192][k=64] ; s[m][k=48][h=192]
  const int total = 1024 * 192 * 64, stride = gridDim.x * blockDim.x;
  for (int i = blockIdx.x * blockDim.x + threadIdx.x; i < total; i += stride) {
    const int m = i / 12288, r = i - m * 12288, h = r >> 6, k = r & 63;
    d[i] = (k < 48) ? (_Float16)s[(m * 48 + k) * 192 + h] : (_Float16)0.f;
  }
}
__global__ void prep_w2c(const float* __restrict__ s, _Float16* __restrict__ d) {
  // d[m][o=48][k=192] ; s[m][k=192][o=48]
  const int total = 1024 * 48 * 192, stride = gridDim.x * blockDim.x;
  for (int i = blockIdx.x * blockDim.x + threadIdx.x; i < total; i += stride) {
    const int m = i / 9216, r = i - m * 9216, o = r / 192, k = r - o * 192;
    d[i] = (_Float16)s[(m * 192 + k) * 48 + o];
  }
}
__global__ void prep_w1v(const float* __restrict__ s, _Float16* __restrict__ d) {
  // d[n][h=112][k=32] ; s[n][k=25][h=100]
  const int total = 2048 * 112 * 32, stride = gridDim.x * blockDim.x;
  for (int i = blockIdx.x * blockDim.x + threadIdx.x; i < total; i += stride) {
    const int n = i / 3584, r = i - n * 3584, h = r >> 5, k = r & 31;
    d[i] = (k < 25 && h < 100) ? (_Float16)s[(n * 25 + k) * 100 + h]
                               : (_Float16)0.f;
  }
}
__global__ void prep_w2v(const float* __restrict__ s, _Float16* __restrict__ d) {
  // d[n][o=32][k=128] ; s[n][k=100][o=25]
  const int total = 2048 * 32 * 128, stride = gridDim.x * blockDim.x;
  for (int i = blockIdx.x * blockDim.x + threadIdx.x; i < total; i += stride) {
    const int n = i >> 12, r = i & 4095, o = r >> 7, k = r & 127;
    d[i] = (k < 100 && o < 25) ? (_Float16)s[(n * 100 + k) * 25 + o]
                               : (_Float16)0.f;
  }
}

// ---------------------------------------------------------------------------
// Check MLP: one block per check m, 8 waves x 16 batch rows = 128 batch.
// X[128x48->64] @ W1[48x192] -> gelu -> @ W2[192x48], * syndrome sign.
// ---------------------------------------------------------------------------
__global__ __launch_bounds__(256)
void chk_mlp_kernel(const _Float16* __restrict__ msgIn,   // bufMsg [B,6144,8]
                    const float*    __restrict__ prior,   // [N]
                    const int*      __restrict__ v2c,     // [M*DC]
                    const int*      __restrict__ nbrs,    // [M*DC]
                    const int*      __restrict__ synd,    // [B,M]
                    const _Float16* __restrict__ w1t,     // [M,192,64] f16
                    const float*    __restrict__ b1,      // [M,192]
                    const _Float16* __restrict__ w2t,     // [M,48,192] f16
                    const float*    __restrict__ b2,      // [M,48]
                    _Float16*       __restrict__ bufY,    // [B,6144,8]
                    int first) {
  __shared__ __align__(16) _Float16 sX[128 * 64];     // X, K padded 48->64
  __shared__ __align__(16) _Float16 sW1[192 * 64];    // W1^T [h][k]
  __shared__ __align__(16) _Float16 sH[128 * 192];    // gelu(H)
  __shared__ __align__(16) _Float16 sW2[48 * 192];    // W2^T [o][k]

  const int m = blockIdx.x;
  const int tid = threadIdx.x;

  // ---- async-stage weights (contiguous b128 copies) -----------------------
  copy_block(w1t + m * 12288, sW1, 12288, tid, 256);
  copy_block(w2t + m * 9216,  sW2, 9216,  tid, 256);

  // ---- stage X ------------------------------------------------------------
  if (first) {
    for (int i = tid; i < 128 * 64; i += 256) {
      const int k = i & 63;
      float v = 0.f;
      if (k < 48 && (k & 7) == 0) v = prior[nbrs[m * 6 + (k >> 3)]];
      sX[i] = (_Float16)v;
    }
  } else {
    // one 16B edge payload (D=8 f16) per (batch row, slot)
    for (int i = tid; i < 128 * 6; i += 256) {
      const int b = i / 6, slot = i - b * 6;
      const int e = v2c[m * 6 + slot];                // check-edge -> var-edge
      g2l_b128(msgIn + b * 49152 + e * 8, sX + b * 64 + slot * 8);
    }
    // zero pad k = 48..63
    for (int i = tid; i < 128 * 2; i += 256) {
      const int b = i >> 1, q = i & 1;
      *(int4*)(sX + b * 64 + 48 + q * 8) = int4{0, 0, 0, 0};
    }
  }
  wait_async_copies();
  __syncthreads();

  const int wave = tid >> 5, lane = tid & 31;
  const int half = lane >> 4, col = lane & 15;
  const int b0 = wave * 16;

  // ---- layer 1: 16x192 strip, K=64 (12 N-tiles x 2 K-steps) ---------------
  const v16h a0 = load_a_frag(sX + b0 * 64, 64, 0, lane);
  const v16h a1 = load_a_frag(sX + b0 * 64, 64, 32, lane);
#pragma unroll
  for (int nt = 0; nt < 12; ++nt) {
    v8f acc = {};
    acc = wmma_f16(a0, load_bt_frag(sW1 + (nt * 16) * 64, 64, 0, lane), acc);
    acc = wmma_f16(a1, load_bt_frag(sW1 + (nt * 16) * 64, 64, 32, lane), acc);
    const float bias = b1[m * 192 + nt * 16 + col];
#pragma unroll
    for (int i = 0; i < 8; ++i) {
      const float g = gelu_fast(acc[i] + bias);
      sH[(b0 + i + 8 * half) * 192 + nt * 16 + col] = (_Float16)g;
    }
  }
  __syncthreads();

  // syndrome sign per accumulator row
  float sg[8];
#pragma unroll
  for (int i = 0; i < 8; ++i)
    sg[i] = (float)(1 - 2 * synd[(b0 + i + 8 * half) * 1024 + m]);

  // ---- layer 2: K=192 (3 N-tiles x 6 K-steps), sign, emit -----------------
  v16h ah[6];
#pragma unroll
  for (int j = 0; j < 6; ++j) ah[j] = load_a_frag(sH + b0 * 192, 192, 32 * j, lane);
#pragma unroll
  for (int nt = 0; nt < 3; ++nt) {
    v8f acc = {};
#pragma unroll
    for (int j = 0; j < 6; ++j)
      acc = wmma_f16(ah[j], load_bt_frag(sW2 + (nt * 16) * 192, 192, 32 * j, lane), acc);
    const float bias = b2[m * 48 + nt * 16 + col];
    const int k = nt * 16 + col;                       // 0..47
#pragma unroll
    for (int i = 0; i < 8; ++i) {
      const int b = b0 + i + 8 * half;
      bufY[b * 49152 + m * 48 + k] = (_Float16)((acc[i] + bias) * sg[i]);
    }
  }
}

// ---------------------------------------------------------------------------
// Variable MLP: one block per variable n, 8 waves x 16 batch rows.
// X[128x25->32] @ W1[25x100->112] -> gelu -> (K 100->128) @ W2[100x25->32].
// ---------------------------------------------------------------------------
__global__ __launch_bounds__(256)
void var_mlp_kernel(const _Float16* __restrict__ bufY,    // [B,6144,8]
                    const float*    __restrict__ prior,   // [N]
                    const int*      __restrict__ c2v,     // [N*DV]
                    const _Float16* __restrict__ w1t,     // [N,112,32] f16
                    const float*    __restrict__ b1,      // [N,100]
                    const _Float16* __restrict__ w2t,     // [N,32,128] f16
                    const float*    __restrict__ b2,      // [N,25]
                    _Float16*       __restrict__ bufMsg,  // [B,6144,8]
                    float*          __restrict__ out,     // [T,B,N]
                    int t) {
  __shared__ __align__(16) _Float16 sX[128 * 32];    // X, K padded 25->32
  __shared__ __align__(16) _Float16 sW1[112 * 32];   // W1^T [h][k]
  __shared__ __align__(16) _Float16 sH[128 * 128];   // gelu(H), K pad -> 128
  __shared__ __align__(16) _Float16 sW2[32 * 128];   // W2^T [o][k]

  const int n = blockIdx.x;
  const int tid = threadIdx.x;

  copy_block(w1t + n * 3584, sW1, 3584, tid, 256);
  copy_block(w2t + n * 4096, sW2, 4096, tid, 256);

  // X: three 16B edge payloads per batch row + prior scalar + zero pad
  for (int i = tid; i < 128 * 3; i += 256) {
    const int b = i / 3, l = i - b * 3;
    const int e = c2v[n * 3 + l];                    // var-edge -> check-edge
    g2l_b128(bufY + b * 49152 + e * 8, sX + b * 32 + l * 8);
  }
  for (int b = tid; b < 128; b += 256) {
    *(int4*)(sX + b * 32 + 24) = int4{0, 0, 0, 0};   // cols 24..31 = 0
    sX[b * 32 + 24] = (_Float16)prior[n];            // col 24 = prior LLR
  }
  wait_async_copies();
  __syncthreads();

  const int wave = tid >> 5, lane = tid & 31;
  const int half = lane >> 4, col = lane & 15;
  const int b0 = wave * 16;

  // ---- layer 1: K=32, 7 N-tiles -------------------------------------------
  const v16h a = load_a_frag(sX + b0 * 32, 32, 0, lane);
#pragma unroll
  for (int nt = 0; nt < 7; ++nt) {
    v8f acc = {};
    acc = wmma_f16(a, load_bt_frag(sW1 + (nt * 16) * 32, 32, 0, lane), acc);
    const int h = nt * 16 + col;
    const float bias = (h < 100) ? b1[n * 100 + h] : 0.f;
#pragma unroll
    for (int i = 0; i < 8; ++i) {
      const float g = (h < 100) ? gelu_fast(acc[i] + bias) : 0.f;
      sH[(b0 + i + 8 * half) * 128 + h] = (_Float16)g;
    }
  }
  // zero pad columns 112..127 of this wave's strip
  for (int i = lane; i < 16 * 2; i += 32) {
    const int r = i >> 1, q = i & 1;
    *(int4*)(sH + (b0 + r) * 128 + 112 + q * 8) = int4{0, 0, 0, 0};
  }
  __syncthreads();

  // ---- layer 2: K=128 (2 N-tiles x 4 K-steps) -----------------------------
  v16h ah[4];
#pragma unroll
  for (int j = 0; j < 4; ++j) ah[j] = load_a_frag(sH + b0 * 128, 128, 32 * j, lane);
#pragma unroll
  for (int nt = 0; nt < 2; ++nt) {
    v8f acc = {};
#pragma unroll
    for (int j = 0; j < 4; ++j)
      acc = wmma_f16(ah[j], load_bt_frag(sW2 + (nt * 16) * 128, 128, 32 * j, lane), acc);
    const int o = nt * 16 + col;                       // 0..31, valid < 25
    const float bias = (o < 25) ? b2[n * 25 + o] : 0.f;
#pragma unroll
    for (int i = 0; i < 8; ++i) {
      const int b = b0 + i + 8 * half;
      const float y = acc[i] + bias;
      if (o < 24) {
        bufMsg[b * 49152 + n * 24 + o] = (_Float16)y;  // messages v->c
      } else if (o == 24) {
        out[((size_t)t * 128 + b) * 2048 + n] = y;     // LLR, fp32
      }
    }
  }
}

// ---------------------------------------------------------------------------
// Launch
// ---------------------------------------------------------------------------
extern "C" void kernel_launch(void* const* d_in, const int* in_sizes, int n_in,
                              void* d_out, int out_size, void* d_ws, size_t ws_size,
                              hipStream_t stream) {
  (void)in_sizes; (void)n_in; (void)out_size; (void)ws_size;

  const int*   synd    = (const int*)  d_in[0];
  const float* prior   = (const float*)d_in[1];
  const float* chk_w1  = (const float*)d_in[2];
  const float* chk_b1  = (const float*)d_in[3];
  const float* chk_w2  = (const float*)d_in[4];
  const float* chk_b2  = (const float*)d_in[5];
  const float* var_w1  = (const float*)d_in[6];
  const float* var_b1  = (const float*)d_in[7];
  const float* var_w2  = (const float*)d_in[8];
  const float* var_b2  = (const float*)d_in[9];
  const int*   nbrs    = (const int*)  d_in[10];
  const int*   c2v     = (const int*)  d_in[11];
  const int*   v2c     = (const int*)  d_in[12];
  float* out = (float*)d_out;

  // workspace carve-up (f16, 256B aligned)
  char* ws = (char*)d_ws;
  auto carve = [&ws](size_t elems) {
    _Float16* p = (_Float16*)ws;
    ws += ((elems * sizeof(_Float16) + 255) / 256) * 256;
    return p;
  };
  const int NW1C = 1024 * 192 * 64;    // transposed, K-padded
  const int NW2C = 1024 * 48 * 192;
  const int NW1V = 2048 * 112 * 32;
  const int NW2V = 2048 * 32 * 128;
  const int NBUF = 128 * 6144 * 8;

  _Float16* w1c    = carve(NW1C);
  _Float16* w2c    = carve(NW2C);
  _Float16* w1v    = carve(NW1V);
  _Float16* w2v    = carve(NW2V);
  _Float16* bufY   = carve(NBUF);
  _Float16* bufMsg = carve(NBUF);

  prep_w1c<<<2048, 256, 0, stream>>>(chk_w1, w1c);
  prep_w2c<<<2048, 256, 0, stream>>>(chk_w2, w2c);
  prep_w1v<<<2048, 256, 0, stream>>>(var_w1, w1v);
  prep_w2v<<<2048, 256, 0, stream>>>(var_w2, w2v);

  for (int t = 0; t < 5; ++t) {
    chk_mlp_kernel<<<1024, 256, 0, stream>>>(
        bufMsg, prior, v2c, nbrs, synd,
        w1c, chk_b1, w2c, chk_b2, bufY, (t == 0) ? 1 : 0);
    var_mlp_kernel<<<2048, 256, 0, stream>>>(
        bufY, prior, c2v,
        w1v, var_b1, w2v, var_b2, bufMsg, out, t);
  }
}
